// SpeechRecognitionModel_50723563766422
// MI455X (gfx1250) — matrix operand order repaired and checked by reading, not compiled, not run on hardware
//
#include <hip/hip_runtime.h>
#include <hip/hip_bf16.h>
#include <math.h>

#define DEV static __device__ __forceinline__

typedef __bf16 bf16_t;
typedef __attribute__((ext_vector_type(16))) __bf16 v16bf;
typedef __attribute__((ext_vector_type(8)))  float  v8f;

// model dims
#define Bsz   16
#define T0    1024
#define F0c   128
#define Tc    512      // time after stride-2 conv
#define Fc    64       // freq after stride-2 conv
#define CC    32       // cnn channels
#define Mrows (Tc*Bsz) // 8192 GEMM rows
#define Hdim  1024
#define Dd    512
#define NSRU  3072     // 2*3*512

DEV bf16_t f2bf(float f) {
  union { float f; unsigned u; } v; v.f = f;
  unsigned r = v.u + 0x7fffu + ((v.u >> 16) & 1u);   // round-to-nearest-even
  unsigned short h = (unsigned short)(r >> 16);
  bf16_t o; __builtin_memcpy(&o, &h, 2);
  return o;
}

DEV float sigm(float x) { return 1.0f / (1.0f + __expf(-x)); }

// ---------------- CNN stage ----------------

// initial 1->32 conv, 3x3, stride 2, pad 1. one thread per output (b,co,f,t)
__global__ void conv0_kernel(const float* __restrict__ x, const float* __restrict__ w,
                             const float* __restrict__ bias, float* __restrict__ out) {
  int idx = blockIdx.x * blockDim.x + threadIdx.x;   // 2^24 total
  int t  = idx & (Tc - 1);
  int f  = (idx >> 9) & (Fc - 1);
  int co = (idx >> 15) & (CC - 1);
  int b  = idx >> 20;
  float acc = bias[co];
  #pragma unroll
  for (int kh = 0; kh < 3; ++kh) {
    int fi = 2 * f + kh - 1;
    if (fi < 0 || fi >= F0c) continue;
    #pragma unroll
    for (int kw = 0; kw < 3; ++kw) {
      int ti = 2 * t + kw - 1;
      if (ti < 0 || ti >= T0) continue;
      acc += x[((size_t)b * F0c + fi) * T0 + ti] * w[co * 9 + kh * 3 + kw];
    }
  }
  out[idx] = acc;
}

// LayerNorm over the F (=64) axis per (b,c,t), scale per-f, then ReLU
__global__ void ln_feats_relu_kernel(const float* __restrict__ X, float* __restrict__ Y,
                                     const float* __restrict__ g, const float* __restrict__ be) {
  int idx = blockIdx.x * blockDim.x + threadIdx.x;   // 16*32*512
  int t = idx & (Tc - 1);
  int c = (idx >> 9) & (CC - 1);
  int b = idx >> 14;
  size_t base = ((size_t)(b * CC + c) * Fc) * Tc + t;
  float v[Fc];
  float m = 0.f;
  #pragma unroll
  for (int f = 0; f < Fc; ++f) { v[f] = X[base + (size_t)f * Tc]; m += v[f]; }
  m *= (1.0f / Fc);
  float var = 0.f;
  #pragma unroll
  for (int f = 0; f < Fc; ++f) { float d = v[f] - m; var += d * d; }
  var *= (1.0f / Fc);
  float inv = rsqrtf(var + 1e-5f);
  #pragma unroll
  for (int f = 0; f < Fc; ++f) {
    float o = (v[f] - m) * inv * g[f] + be[f];
    Y[base + (size_t)f * Tc] = o > 0.f ? o : 0.f;
  }
}

// 32->32 3x3 conv, pad 1, stride 1. weights staged in LDS. optional residual add.
__global__ void conv3x3_kernel(const float* __restrict__ X, const float* __restrict__ W,
                               const float* __restrict__ bias, const float* __restrict__ res,
                               float* __restrict__ out) {
  __shared__ float wl[CC * CC * 9];    // 36 KB
  for (int i = threadIdx.x; i < CC * CC * 9; i += blockDim.x) wl[i] = W[i];
  __syncthreads();
  int idx = blockIdx.x * blockDim.x + threadIdx.x;   // (b,co,f,t), 2^24
  int t  = idx & (Tc - 1);
  int f  = (idx >> 9) & (Fc - 1);
  int co = (idx >> 15) & (CC - 1);
  int b  = idx >> 20;
  float acc = bias[co];
  for (int ci = 0; ci < CC; ++ci) {
    size_t ibase = ((size_t)(b * CC + ci) * Fc) * Tc;
    const float* wp = &wl[(co * CC + ci) * 9];
    #pragma unroll
    for (int kh = 0; kh < 3; ++kh) {
      int fi = f + kh - 1;
      if (fi < 0 || fi >= Fc) continue;
      #pragma unroll
      for (int kw = 0; kw < 3; ++kw) {
        int ti = t + kw - 1;
        if (ti < 0 || ti >= Tc) continue;
        acc += X[ibase + (size_t)fi * Tc + ti] * wp[kh * 3 + kw];
      }
    }
  }
  if (res) acc += res[idx];
  out[idx] = acc;
}

// ---------------- WMMA GEMM path ----------------
// CDNA5 16-bit A (16x32) fragment k-offset: lane half in {0,1}, q = 0..15 in-lane element
DEV int koff_a(int q, int hf) { int j = q >> 1, s = q & 1;
  return ((j >> 2) << 4) + ((j & 3) << 1) + (hf << 3) + s; }
// B (32x16): same k order as A: q<8 -> K=8*hf+q, q>=8 -> K=16+8*hf+(q-8)
DEV int koff_b(int q, int hf) { return ((q >> 3) << 4) + (hf << 3) + (q & 7); }

// pack fp32 row-major A[M,K] -> fragment-linear bf16 tiles:
// tile (mt,kt) = 512 bf16: element = lane*16 + q
__global__ void pack_a_kernel(const float* __restrict__ A, bf16_t* __restrict__ out, int K) {
  int idx = blockIdx.x * blockDim.x + threadIdx.x;
  int Kt = K >> 5;
  int e = idx & 511, tile = idx >> 9;
  int kt = tile % Kt, mt = tile / Kt;
  int lane = e >> 4, q = e & 15;
  int m = (mt << 4) + (lane & 15);
  int k = (kt << 5) + koff_a(q, lane >> 4);
  out[idx] = f2bf(A[(size_t)m * K + k]);
}

// pack projection input directly from CNN tensor: A[m=t*16+b][k=c*64+f] = Xc[b][c][f][t]
__global__ void pack_a_proj_kernel(const float* __restrict__ Xc, bf16_t* __restrict__ out) {
  int idx = blockIdx.x * blockDim.x + threadIdx.x;   // 512 * 64 * 512 = 2^24
  int e = idx & 511, tile = idx >> 9;
  int kt = tile & 63, mt = tile >> 6;                // Kt = 2048/32 = 64
  int lane = e >> 4, q = e & 15;
  int m = (mt << 4) + (lane & 15);
  int k = (kt << 5) + koff_a(q, lane >> 4);
  int t = m >> 4, b = m & 15;
  int c = k >> 6, f = k & 63;
  out[idx] = f2bf(Xc[((size_t)(b * CC + c) * Fc + f) * Tc + t]);
}

// pack fp32 row-major W[K,Nreal] -> B-fragment tiles (N padded with zeros to 16-multiples)
__global__ void pack_w_kernel(const float* __restrict__ W, bf16_t* __restrict__ out,
                              int K, int Nreal) {
  int idx = blockIdx.x * blockDim.x + threadIdx.x;
  int Kt = K >> 5;
  int e = idx & 511, tile = idx >> 9;
  int kt = tile % Kt, nt = tile / Kt;
  int lane = e >> 4, q = e & 15;
  int n = (nt << 4) + (lane & 15);
  int k = (kt << 5) + koff_b(q, lane >> 4);
  float v = (n < Nreal) ? W[(size_t)k * Nreal + n] : 0.f;
  out[idx] = f2bf(v);
}

// C[M,Npad] = A @ W (+bias)(+relu). block = 4 waves; wave tile 64x64 = 4x4 WMMA accums.
// __launch_bounds__(128, 1): budget VGPRs for a single wave/SIMD so the 16
// f32 accumulators (128 VGPRs) + fragments stay resident (no scratch spills).
__global__ __launch_bounds__(128, 1) void gemm_wmma_kernel(
                                 const bf16_t* __restrict__ Ap, const bf16_t* __restrict__ Bp,
                                 float* __restrict__ C, int Kt, int Npad,
                                 const float* __restrict__ bias, int Nreal, int relu) {
  int lane = threadIdx.x & 31;
  int wv   = threadIdx.x >> 5;
  int mt0  = blockIdx.x << 2;
  int nt0  = (blockIdx.y << 4) + (wv << 2);
  if ((nt0 << 4) >= Npad) return;
  v8f acc[4][4];
  #pragma unroll
  for (int i = 0; i < 4; ++i)
    #pragma unroll
    for (int j = 0; j < 4; ++j) { v8f z = {}; acc[i][j] = z; }
  const int lo = lane << 4;
  const bf16_t* aptr = Ap + ((size_t)mt0 * Kt << 9) + lo;
  const bf16_t* bptr = Bp + ((size_t)nt0 * Kt << 9) + lo;
  const size_t tileStride = (size_t)Kt << 9;   // elements per 16-row fragment strip
  for (int kt = 0; kt < Kt; ++kt) {
    v16bf a[4];
    #pragma unroll
    for (int i = 0; i < 4; ++i)
      a[i] = *(const v16bf*)(aptr + i * tileStride + ((size_t)kt << 9));
    #pragma unroll
    for (int j = 0; j < 4; ++j) {
      v16bf bfr = *(const v16bf*)(bptr + j * tileStride + ((size_t)kt << 9));
      #pragma unroll
      for (int i = 0; i < 4; ++i)
        acc[i][j] = __builtin_amdgcn_wmma_f32_16x16x32_bf16(
            false, a[i], false, bfr, (short)0, acc[i][j], false, false);
    }
  }
  int hf = lane >> 4, col = lane & 15;
  #pragma unroll
  for (int j = 0; j < 4; ++j) {
    int n = ((nt0 + j) << 4) + col;
    float bv = (bias && n < Nreal) ? bias[n] : 0.f;
    #pragma unroll
    for (int i = 0; i < 4; ++i) {
      #pragma unroll
      for (int r = 0; r < 8; ++r) {
        int m = ((mt0 + i) << 4) + (hf << 3) + r;
        float vv = acc[i][j][r] + bv;
        if (relu && vv < 0.f) vv = 0.f;
        C[(size_t)m * Npad + n] = vv;
      }
    }
  }
}

// ---------------- row LayerNorm (over last dim) ----------------
__global__ void ln_rows_kernel(const float* __restrict__ X, float* __restrict__ Y,
                               const float* __restrict__ g, const float* __restrict__ be, int H) {
  __shared__ float ss[256], sq[256];
  int row = blockIdx.x, tid = threadIdx.x;
  const float* xr = X + (size_t)row * H;
  float s = 0.f, q = 0.f;
  for (int i = tid; i < H; i += 256) { float v = xr[i]; s += v; q += v * v; }
  ss[tid] = s; sq[tid] = q;
  __syncthreads();
  for (int o = 128; o > 0; o >>= 1) {
    if (tid < o) { ss[tid] += ss[tid + o]; sq[tid] += sq[tid + o]; }
    __syncthreads();
  }
  float mean = ss[0] / H;
  float var  = sq[0] / H - mean * mean;
  float inv  = rsqrtf(var + 1e-5f);
  float* yr = Y + (size_t)row * H;
  for (int i = tid; i < H; i += 256) yr[i] = (xr[i] - mean) * inv * g[i] + be[i];
}

// ---------------- SRU recurrence: one thread per (b,dir,d) ----------------
__global__ void sru_scan_kernel(const float* __restrict__ U, const float* __restrict__ X,
                                const float* __restrict__ vv, const float* __restrict__ bb,
                                float* __restrict__ Xout) {
  int idx = blockIdx.x * blockDim.x + threadIdx.x;   // 16*2*512 = 16384
  int d   = idx & (Dd - 1);
  int dir = (idx >> 9) & 1;
  int b   = idx >> 10;
  float v0 = vv[(dir * 2 + 0) * Dd + d];
  float v1 = vv[(dir * 2 + 1) * Dd + d];
  float b0 = bb[(dir * 2 + 0) * Dd + d];
  float b1 = bb[(dir * 2 + 1) * Dd + d];
  int nb = dir * 3 * Dd;
  float c = 0.f;
  for (int s = 0; s < Tc; ++s) {
    int t = dir ? (Tc - 1 - s) : s;
    size_t row = (size_t)(t * Bsz + b);
    size_t ub  = row * NSRU + nb + d;
    float u0 = U[ub];
    float fp = U[ub + Dd] + b0;
    float rp = U[ub + 2 * Dd] + b1;
    float xr = X[row * Hdim + dir * Dd + d];
    float f = sigm(fp + v0 * c);
    float r = sigm(rp + v1 * c);
    c = f * c + (1.f - f) * u0;
    float h = r * tanhf(c) + (1.f - r) * xr;
    Xout[row * Hdim + dir * Dd + d] = h;
  }
}

// d_out[b][t][v<29] = C2[(t*16+b)*64 + v]   (bias already fused in GEMM)
__global__ void final_out_kernel(const float* __restrict__ C2, float* __restrict__ out) {
  int idx = blockIdx.x * blockDim.x + threadIdx.x;
  if (idx >= Bsz * Tc * 29) return;
  int v = idx % 29;
  int t = (idx / 29) % Tc;
  int b = idx / (29 * Tc);
  out[idx] = C2[(size_t)(t * Bsz + b) * 64 + v];
}

extern "C" void kernel_launch(void* const* d_in, const int* in_sizes, int n_in,
                              void* d_out, int out_size, void* d_ws, size_t ws_size,
                              hipStream_t stream) {
  (void)in_sizes; (void)n_in; (void)out_size;
  const float* x       = (const float*)d_in[0];
  const float* cnn_w   = (const float*)d_in[1];
  const float* cnn_b   = (const float*)d_in[2];
  const float* res_w1  = (const float*)d_in[3];
  const float* res_b1  = (const float*)d_in[4];
  const float* res_w2  = (const float*)d_in[5];
  const float* res_b2  = (const float*)d_in[6];
  const float* ln1g    = (const float*)d_in[7];
  const float* ln1b    = (const float*)d_in[8];
  const float* ln2g    = (const float*)d_in[9];
  const float* ln2b    = (const float*)d_in[10];
  const float* proj_w  = (const float*)d_in[11];
  const float* sru_w   = (const float*)d_in[12];
  const float* sru_v   = (const float*)d_in[13];
  const float* sru_b   = (const float*)d_in[14];
  const float* sru_lng = (const float*)d_in[15];
  const float* sru_lnb = (const float*)d_in[16];
  const float* cls_lng = (const float*)d_in[17];
  const float* cls_lnb = (const float*)d_in[18];
  const float* cls_w1  = (const float*)d_in[19];
  const float* cls_b1  = (const float*)d_in[20];
  const float* cls_w2  = (const float*)d_in[21];
  const float* cls_b2  = (const float*)d_in[22];

  const size_t MB = 1024ull * 1024ull;
  if (ws_size < 264ull * MB) return;   // workspace layout needs 264 MiB
  char* ws = (char*)d_ws;
  // phase-overlapped regions (stream-serialized lifetimes):
  float*  cnn0 = (float*)(ws + 0);          // [0,64M)   CNN ping
  float*  cnn1 = (float*)(ws + 64 * MB);    // [64,128M) CNN temp
  float*  cnn2 = (float*)(ws + 128 * MB);   // [128,192M) CNN pong / final CNN X
  float*  U    = (float*)(ws + 0);          // [0,96M)   SRU pre-activations (after CNN dead)
  float*  XL   = (float*)(ws + 96 * MB);    // [96,128M) LN output
  float*  XB   = (float*)(ws + 128 * MB);   // [128,160M) SRU ping-pong (after proj pack)
  float*  C1   = (float*)(ws + 160 * MB);   // [160,176M) cls hidden
  float*  C2   = (float*)(ws + 176 * MB);   // [176,178M) cls logits (padded N=64)
  bf16_t* Ap   = (bf16_t*)(ws + 192 * MB);  // [192,224M) packed A fragments
  bf16_t* Wp   = (bf16_t*)(ws + 224 * MB);  // [224,232M) packed W fragments
  float*  XA   = (float*)(ws + 232 * MB);   // [232,264M) SRU ping-pong

  dim3 blk(256);
  // ---- CNN front-end ----
  conv0_kernel<<<(1 << 24) / 256, blk, 0, stream>>>(x, cnn_w, cnn_b, cnn0);
  float* Xc = cnn0; float* Alt = cnn2;
  for (int i = 0; i < 3; ++i) {
    ln_feats_relu_kernel<<<(Bsz * CC * Tc) / 256, blk, 0, stream>>>(Xc, cnn1, ln1g + i * Fc, ln1b + i * Fc);
    conv3x3_kernel<<<(1 << 24) / 256, blk, 0, stream>>>(cnn1, res_w1 + i * CC * CC * 9, res_b1 + i * CC, nullptr, Alt);
    ln_feats_relu_kernel<<<(Bsz * CC * Tc) / 256, blk, 0, stream>>>(Alt, cnn1, ln2g + i * Fc, ln2b + i * Fc);
    conv3x3_kernel<<<(1 << 24) / 256, blk, 0, stream>>>(cnn1, res_w2 + i * CC * CC * 9, res_b2 + i * CC, Xc, Alt);
    float* tmp = Xc; Xc = Alt; Alt = tmp;  // final Xc = cnn2
  }
  // ---- projection GEMM: (8192 x 2048) @ (2048 x 1024) -> XA ----
  pack_w_kernel<<<(64 * 64 * 512) / 256, blk, 0, stream>>>(proj_w, Wp, 2048, 1024);
  pack_a_proj_kernel<<<(1 << 24) / 256, blk, 0, stream>>>(Xc, Ap);
  gemm_wmma_kernel<<<dim3(Mrows / 64, 1024 / 256), dim3(128), 0, stream>>>(
      Ap, Wp, XA, 2048 / 32, 1024, nullptr, 1024, 0);

  // ---- 3 bidirectional SRU layers ----
  float* cur = XA; float* oth = XB;
  for (int l = 0; l < 3; ++l) {
    ln_rows_kernel<<<Mrows, blk, 0, stream>>>(cur, XL, sru_lng + l * Hdim, sru_lnb + l * Hdim, Hdim);
    pack_a_kernel<<<(Mrows / 16) * (Hdim / 32) * 512 / 256, blk, 0, stream>>>(XL, Ap, Hdim);
    pack_w_kernel<<<(NSRU / 16) * (Hdim / 32) * 512 / 256, blk, 0, stream>>>(
        sru_w + (size_t)l * Hdim * NSRU, Wp, Hdim, NSRU);
    gemm_wmma_kernel<<<dim3(Mrows / 64, NSRU / 256), dim3(128), 0, stream>>>(
        Ap, Wp, U, Hdim / 32, NSRU, nullptr, NSRU, 0);
    sru_scan_kernel<<<(Bsz * 2 * Dd) / 256, blk, 0, stream>>>(
        U, cur, sru_v + l * 4 * Dd, sru_b + l * 4 * Dd, oth);
    float* t2 = cur; cur = oth; oth = t2;  // after 3 layers cur = XB
  }

  // ---- classifier ----
  ln_rows_kernel<<<Mrows, blk, 0, stream>>>(cur, XL, cls_lng, cls_lnb, Hdim);
  pack_a_kernel<<<(Mrows / 16) * (Hdim / 32) * 512 / 256, blk, 0, stream>>>(XL, Ap, Hdim);
  pack_w_kernel<<<(512 / 16) * (Hdim / 32) * 512 / 256, blk, 0, stream>>>(cls_w1, Wp, Hdim, 512);
  gemm_wmma_kernel<<<dim3(Mrows / 64, 512 / 256), dim3(128), 0, stream>>>(
      Ap, Wp, C1, Hdim / 32, 512, cls_b1, 512, 1);
  pack_a_kernel<<<(Mrows / 16) * (512 / 32) * 512 / 256, blk, 0, stream>>>(C1, Ap, 512);
  pack_w_kernel<<<(64 / 16) * (512 / 32) * 512 / 256, blk, 0, stream>>>(cls_w2, Wp, 512, 29);
  gemm_wmma_kernel<<<dim3(Mrows / 64, 1), dim3(128), 0, stream>>>(
      Ap, Wp, C2, 512 / 32, 64, cls_b2, 29, 0);
  final_out_kernel<<<(Bsz * Tc * 29 + 255) / 256, blk, 0, stream>>>(C2, (float*)d_out);
}
